// ScaledDotProductAttention_75041668596041
// MI455X (gfx1250) — matrix operand order repaired and checked by reading, not compiled, not run on hardware
//
#include <hip/hip_runtime.h>

// Problem constants (match the reference)
constexpr int kB = 4, kH = 16, kS = 2048, kD = 64;
constexpr int kBH = kB * kH;
constexpr float kScale = 0.125f; // 1/sqrt(64)

constexpr int BLOCK_M = 64;    // query rows per workgroup
constexpr int BLOCK_N = 128;   // key rows per iteration
constexpr int NTHREADS = 128;  // 4 waves (wave32)

// LDS strides (element units). Multiples of 8 halves / 4 floats keep 16B
// alignment for b128 vector ops; odd-ish multiples spread banks.
constexpr int K_STRIDE  = 72;  // halves, Ks[key][d]        (144B rows)
constexpr int VT_STRIDE = 136; // halves, Vt[d][key]        (272B rows)
constexpr int S_STRIDE  = 132; // floats, Ss[row][key]      (528B rows)
constexpr int P_STRIDE  = 136; // halves, Ps[row][key]      (272B rows)

typedef __attribute__((ext_vector_type(16))) _Float16 v16h;
typedef __attribute__((ext_vector_type(4)))  _Float16 v4h;
typedef __attribute__((ext_vector_type(8)))  float    v8f;
typedef __attribute__((ext_vector_type(4)))  float    v4f;

// A-matrix (16x32 f16) k index for element i, lane-half h (ISA 7.12.2):
//   i<8  -> h*8 + i          (K 0..7  / 8..15)
//   i>=8 -> h*8 + (i-8) + 16 (K 16..23 / 24..31)
__device__ __forceinline__ int ka_idx(int i, int h) {
  return h * 8 + (i & 7) + ((i >> 3) << 4);
}
// B-matrix (32x16 f16) k index for element i, lane-half h: same order as A
__device__ __forceinline__ int kb_idx(int i, int h) {
  return h * 8 + (i & 7) + ((i >> 3) << 4);
}

__launch_bounds__(NTHREADS)
__global__ void fa_causal_wmma_kernel(const float* __restrict__ Q,
                                      const float* __restrict__ K,
                                      const float* __restrict__ V,
                                      float* __restrict__ O)
{
  __shared__ _Float16 Ks[BLOCK_N * K_STRIDE];   // K block, [key][d], f16
  __shared__ _Float16 Vt[kD * VT_STRIDE];       // V block transposed, [d][key]
  __shared__ float    Ss[BLOCK_M * S_STRIDE];   // raw scores
  __shared__ _Float16 Ps[BLOCK_M * P_STRIDE];   // softmax probs, f16
  __shared__ float    alpha_s[BLOCK_M];
  __shared__ float    invl_s[BLOCK_M];

  const int tid    = threadIdx.x;
  const int wave   = tid >> 5;
  const int lane   = tid & 31;
  const int lane15 = lane & 15;
  const int half   = (lane >> 4) & 1;   // lane-half (0: lanes 0-15, 1: 16-31)
  const int rowbase = half * 8;         // C/D fragment row base for this half

  const int qblk = blockIdx.x;
  const int bh   = blockIdx.y;
  const int q0   = qblk * BLOCK_M;

  const float* Qb = Q + (size_t)bh * kS * kD;
  const float* Kb = K + (size_t)bh * kS * kD;
  const float* Vb = V + (size_t)bh * kS * kD;
  float*       Ob = O + (size_t)bh * kS * kD;

  // ---- Preload this wave's Q rows as two A fragments (d 0..31, 32..63) ----
  const int qrow = q0 + wave * 16 + lane15;  // A layout: M = lane15
  v16h qfrag[2];
#pragma unroll
  for (int c = 0; c < 2; ++c) {
    const float* qp = Qb + (size_t)qrow * kD + c * 32;
#pragma unroll
    for (int i = 0; i < 16; ++i)
      qfrag[c][i] = (_Float16)qp[ka_idx(i, half)];
  }

  // ---- O accumulators: 4 C fragments (16 rows x 64 d per wave) ----
  v8f ofrag[4];
#pragma unroll
  for (int t = 0; t < 4; ++t)
#pragma unroll
    for (int r = 0; r < 8; ++r) ofrag[t][r] = 0.0f;

  // Online-softmax state: 2 threads per row (row = tid>>1), replicated in pair
  float m_i = -3.0e38f;
  float l_i = 0.0f;

  // V-transpose staging role: fixed d column, two groups of 4-key chunks
  const int vd = tid & 63;       // d (0..63)
  const int vg = tid >> 6;       // 0 or 1

  // Causal: key blocks 0 .. floor(q0/BLOCK_N); only the last crosses diagonal
  const int nblocks = (q0 >> 7) + 1;
  for (int jb = 0; jb < nblocks; ++jb) {
    const int j = jb * BLOCK_N;
    const bool diag = (jb == nblocks - 1);

    // Prefetch next K/V block toward L2 while we work on this one
    if (jb + 1 < nblocks) {
      const float* nk = Kb + (size_t)(j + BLOCK_N) * kD;
      const float* nv = Vb + (size_t)(j + BLOCK_N) * kD;
      __builtin_prefetch(nk + (size_t)tid * 64, 0, 1);
      __builtin_prefetch(nv + (size_t)tid * 64, 0, 1);
    }

    // ---- Stage K block: float4 global loads -> v4h b64 LDS stores ----
#pragma unroll
    for (int e = tid; e < BLOCK_N * kD / 4; e += NTHREADS) {
      const int r  = e >> 4;         // key row (16 float4 per row)
      const int c4 = (e & 15) * 4;   // d
      const v4f x = *(const v4f*)(Kb + (size_t)(j + r) * kD + c4);
      v4h hx;
#pragma unroll
      for (int u = 0; u < 4; ++u) hx[u] = (_Float16)x[u];
      *(v4h*)&Ks[r * K_STRIDE + c4] = hx;
    }

    // ---- Stage V transposed: Vt[d][key]; 4 keys per b64 store ----
#pragma unroll
    for (int kg = vg; kg < BLOCK_N / 4; kg += 2) {
      const int key0 = 4 * kg;
      v4h hv;
#pragma unroll
      for (int u = 0; u < 4; ++u)
        hv[u] = (_Float16)Vb[(size_t)(j + key0 + u) * kD + vd];
      *(v4h*)&Vt[vd * VT_STRIDE + key0] = hv;
    }
    __syncthreads();

    // ---- S = (Q K^T) * scale for this wave's 16 rows x 128 keys ----
#pragma unroll
    for (int nt = 0; nt < 8; ++nt) {
      v8f c;
#pragma unroll
      for (int r = 0; r < 8; ++r) c[r] = 0.0f;
#pragma unroll
      for (int kc = 0; kc < 2; ++kc) {
        // B[k=d, n=key]: lane holds key = nt*16+lane15; halves 8h..8h+7 and 16+8h..16+8h+7
        const _Float16* kp = &Ks[(nt * 16 + lane15) * K_STRIDE + kc * 32];
        v16h b;
#pragma unroll
        for (int i = 0; i < 16; ++i) b[i] = kp[kb_idx(i, half)];
        c = __builtin_amdgcn_wmma_f32_16x16x32_f16(
                false, qfrag[kc], false, b, (short)0, c, false, false);
      }
      // scale + causal mask + spill (C layout: n = lane15, m = rowbase+r)
      const int keyg = j + nt * 16 + lane15;
#pragma unroll
      for (int r = 0; r < 8; ++r) {
        const int mrow = wave * 16 + rowbase + r;      // row in block
        float s = c[r] * kScale;
        if (diag && keyg > q0 + mrow) s = -1.0e30f;
        Ss[mrow * S_STRIDE + nt * 16 + lane15] = s;
      }
    }
    __syncthreads();

    // ---- Online softmax: 2 threads per row, combine via shfl_xor(1) ----
    {
      const int srow_id = tid >> 1;      // row 0..63
      const int shalf   = tid & 1;       // which 64-key half
      const float* srow = &Ss[srow_id * S_STRIDE + shalf * 64];
      float mx = m_i;
#pragma unroll 8
      for (int n = 0; n < 64; ++n) mx = fmaxf(mx, srow[n]);
      mx = fmaxf(mx, __shfl_xor(mx, 1, 32));      // pair-combined row max
      const float alpha = __expf(m_i - mx);
      float sum = 0.0f;
      _Float16* prow = &Ps[srow_id * P_STRIDE + shalf * 64];
#pragma unroll 8
      for (int n = 0; n < 64; ++n) {
        const float p = __expf(srow[n] - mx);
        sum += p;
        prow[n] = (_Float16)p;
      }
      sum += __shfl_xor(sum, 1, 32);              // pair-combined row sum
      l_i = alpha * l_i + sum;
      m_i = mx;
      if (shalf == 0) alpha_s[srow_id] = alpha;
    }
    __syncthreads();

    // ---- Rescale O accumulators by per-row alpha ----
    float av[8];
#pragma unroll
    for (int r = 0; r < 8; ++r) av[r] = alpha_s[wave * 16 + rowbase + r];
#pragma unroll
    for (int t = 0; t < 4; ++t)
#pragma unroll
      for (int r = 0; r < 8; ++r) ofrag[t][r] *= av[r];

    // ---- O += P V: A = P rows (contiguous b128), B = Vt rows (contiguous) ----
#pragma unroll
    for (int kc = 0; kc < 4; ++kc) {
      const _Float16* pp = &Ps[(wave * 16 + lane15) * P_STRIDE + kc * 32];
      v16h a;
#pragma unroll
      for (int i = 0; i < 16; ++i) a[i] = pp[ka_idx(i, half)];
#pragma unroll
      for (int nt = 0; nt < 4; ++nt) {
        // B[k=key, n=d]: lane holds d = nt*16+lane15; keys contiguous in Vt
        const _Float16* vp = &Vt[(nt * 16 + lane15) * VT_STRIDE + kc * 32];
        v16h b;
#pragma unroll
        for (int i = 0; i < 16; ++i) b[i] = vp[kb_idx(i, half)];
        ofrag[nt] = __builtin_amdgcn_wmma_f32_16x16x32_f16(
                        false, a, false, b, (short)0, ofrag[nt], false, false);
      }
    }
    __syncthreads(); // protect Ks/Vt/Ps/alpha before next iteration's stores
  }

  // ---- Finalize: divide by running sum, write O (f32) ----
  if ((tid & 1) == 0) invl_s[tid >> 1] = 1.0f / l_i;
  __syncthreads();
  float iv[8];
#pragma unroll
  for (int r = 0; r < 8; ++r) iv[r] = invl_s[wave * 16 + rowbase + r];
#pragma unroll
  for (int nt = 0; nt < 4; ++nt) {
#pragma unroll
    for (int r = 0; r < 8; ++r) {
      const int mrow = wave * 16 + rowbase + r;
      Ob[(size_t)(q0 + mrow) * kD + nt * 16 + lane15] = ofrag[nt][r] * iv[r];
    }
  }
}

extern "C" void kernel_launch(void* const* d_in, const int* in_sizes, int n_in,
                              void* d_out, int out_size, void* d_ws, size_t ws_size,
                              hipStream_t stream) {
  (void)in_sizes; (void)n_in; (void)out_size; (void)d_ws; (void)ws_size;
  const float* q = (const float*)d_in[0];
  const float* k = (const float*)d_in[1];
  const float* v = (const float*)d_in[2];
  // d_in[3] is the causal mask; it is regenerated analytically in-kernel.
  float* o = (float*)d_out;

  dim3 grid(kS / BLOCK_M, kBH);
  dim3 block(NTHREADS);
  hipLaunchKernelGGL(fa_causal_wmma_kernel, grid, block, 0, stream, q, k, v, o);
}
